// ComRoPEAttention_27573690041019
// MI455X (gfx1250) — matrix operand order repaired and checked by reading, not compiled, not run on hardware
//
#include <hip/hip_runtime.h>
#include <hip/hip_bf16.h>

typedef __attribute__((ext_vector_type(16))) _Float16 v16h;
typedef __attribute__((ext_vector_type(8)))  _Float16 v8h;
typedef __attribute__((ext_vector_type(8)))  float    v8f;

#define BATCH 2
#define SEQ   2048
#define HID   1024
#define HN    16
#define DH    64
#define MROWS (BATCH * SEQ)   // 4096

#define NEG_INF (-__builtin_huge_valf())

// Fused-attention LDS layout (floats), carved from dynamic shared memory:
//   sc    : 16 x 2048 score strip              = 32768
//   rstat : 16 x 17 per-row chunk partials     =   272
//   rowst : rowmax[16] + rowinv[16]            =    32
//   pacc  : 8 waves x 32 lanes x 8 accum       =  2048
#define SM_SC    0
#define SM_RSTAT (SM_SC + 16 * SEQ)
#define SM_ROWST (SM_RSTAT + 16 * 17)
#define SM_PACC  (SM_ROWST + 32)
#define SM_FLOATS (SM_PACC + 8 * 32 * 8)
#define SM_BYTES  (SM_FLOATS * 4)               // 140480 bytes < 320 KB WGP LDS

// ---------------------------------------------------------------------------
// WMMA fragment builders (CDNA5 16x16x32 f16 layouts, wave32)
// A (16x32, 16-bit): lanes 0-15 -> M=0..15 holding K = 0..7,16..23;
//                    lanes 16-31 -> M=0..15 holding K = 8..15,24..31.
// B (32x16, 16-bit): lane holds column N=lane&15; lanes 0-15 K = 0..7,16..23;
//                    lanes 16-31 K = 8..15,24..31 (same order as A).
// C/D (16x16, f32):  VGPR r, lanes 0-15 -> M=r, lanes 16-31 -> M=8+r; N=lane&15.
// ---------------------------------------------------------------------------

__device__ __forceinline__ v16h frag_a_f32(const float* A, int lda, int row,
                                           int kk, int lane) {
    int kb = kk + ((lane >> 4) << 3);
    const float* p = A + (size_t)row * lda + kb;
    v16h r;
#pragma unroll
    for (int i = 0; i < 8; ++i) {
        r[i]     = (_Float16)p[i];
        r[i + 8] = (_Float16)p[i + 16];
    }
    return r;
}

__device__ __forceinline__ v16h frag_a_f16(const _Float16* A, int lda, int row,
                                           int kk, int lane) {
    int kb = kk + ((lane >> 4) << 3);
    const _Float16* p = A + (size_t)row * lda + kb;
    v8h lo = *(const v8h*)p;
    v8h hi = *(const v8h*)(p + 16);
    v16h r;
#pragma unroll
    for (int i = 0; i < 8; ++i) { r[i] = lo[i]; r[i + 8] = hi[i]; }
    return r;
}

// A fragment sourced from an LDS f32 strip (row-major, stride SEQ)
__device__ __forceinline__ v16h frag_a_lds(const float* sc, int row,
                                           int kk, int lane) {
    int kb = kk + ((lane >> 4) << 3);
    const float* p = sc + row * SEQ + kb;
    v16h r;
#pragma unroll
    for (int i = 0; i < 8; ++i) {
        r[i]     = (_Float16)p[i];
        r[i + 8] = (_Float16)p[i + 16];
    }
    return r;
}

// B[k][n] = Wt[n][k]  (row-major matrix holding the transposed operand)
__device__ __forceinline__ v16h frag_b_f32T(const float* Wt, int ldb, int col,
                                            int kk, int lane) {
    int kb = kk + ((lane >> 4) << 3);
    const float* p = Wt + (size_t)col * ldb + kb;
    v16h r;
#pragma unroll
    for (int i = 0; i < 8; ++i) { r[i] = (_Float16)p[i]; r[i + 8] = (_Float16)p[i + 16]; }
    return r;
}

__device__ __forceinline__ v16h frag_b_f16T(const _Float16* Bt, int ldb, int col,
                                            int kk, int lane) {
    int kb = kk + ((lane >> 4) << 3);
    const _Float16* p = Bt + (size_t)col * ldb + kb;
    v8h lo = *(const v8h*)p;
    v8h hi = *(const v8h*)(p + 16);
    v16h r;
#pragma unroll
    for (int i = 0; i < 8; ++i) { r[i] = lo[i]; r[i + 8] = hi[i]; }
    return r;
}

__device__ __forceinline__ v8f wmma_f16(v16h a, v16h b, v8f c) {
    return __builtin_amdgcn_wmma_f32_16x16x32_f16(false, a, false, b,
                                                  (short)0, c, false, false);
}

// ---------------------------------------------------------------------------
// 1. shift[s] = mean(ctx_bias[s, :S]) * 0.1
// ---------------------------------------------------------------------------
__global__ void shift_kernel(const float* __restrict__ cbias,
                             float* __restrict__ shift) {
    __shared__ float red[256];
    int s = blockIdx.x;
    const float* row = cbias + (size_t)s * SEQ;
    float acc = 0.f;
    for (int j = threadIdx.x; j < SEQ; j += 256) acc += row[j];
    red[threadIdx.x] = acc;
    __syncthreads();
    for (int o = 128; o > 0; o >>= 1) {
        if ((int)threadIdx.x < o) red[threadIdx.x] += red[threadIdx.x + o];
        __syncthreads();
    }
    if (threadIdx.x == 0) shift[s] = red[0] * (0.1f / (float)SEQ);
}

// ---------------------------------------------------------------------------
// 2. Fused QKV projection: raw[m,n] = sum_k x[m,k] * W[n,k]
//    Register-blocked: one wave -> 16x64 output (A frag reused across 4 WMMAs)
// ---------------------------------------------------------------------------
__global__ void proj_kernel(const float* __restrict__ x,
                            const float* __restrict__ Wq,
                            const float* __restrict__ Wk,
                            const float* __restrict__ Wv,
                            float* __restrict__ qraw,
                            float* __restrict__ kraw,
                            float* __restrict__ vraw) {
    int wave = blockIdx.x * (blockDim.x >> 5) + (threadIdx.x >> 5);
    int lane = threadIdx.x & 31;
    const int ngrp   = HID / 64;                   // 16 groups of 4 n-tiles
    const int mtiles = MROWS / 16;                 // 256
    int mat = wave / (mtiles * ngrp);
    int rem = wave % (mtiles * ngrp);
    int mt = rem / ngrp, ng = rem % ngrp;
    const float* W  = (mat == 0) ? Wq : (mat == 1) ? Wk : Wv;
    float*       Od = (mat == 0) ? qraw : (mat == 1) ? kraw : vraw;
    int m0 = mt * 16, n0 = ng * 64;
    int arow = m0 + (lane & 15);
    v8f acc[4] = {};
    for (int kk = 0; kk < HID; kk += 32) {
        v16h a = frag_a_f32(x, HID, arow, kk, lane);
#pragma unroll
        for (int t = 0; t < 4; ++t) {
            v16h b = frag_b_f32T(W, HID, n0 + t * 16 + (lane & 15), kk, lane);
            acc[t] = wmma_f16(a, b, acc[t]);
        }
    }
    int orow = m0 + ((lane >> 4) << 3);
#pragma unroll
    for (int t = 0; t < 4; ++t) {
        int ocol = n0 + t * 16 + (lane & 15);
#pragma unroll
        for (int r = 0; r < 8; ++r)
            Od[(size_t)(orow + r) * HID + ocol] = acc[t][r];
    }
}

// ---------------------------------------------------------------------------
// 3. RoPE + shift, layout change, f16 down-convert.  q,k -> [B,H,S,D] f16;
//    v -> transposed [B,H,D,S] f16 (so attn@V B-fragments are contiguous).
// ---------------------------------------------------------------------------
__global__ void rope_kernel(const float* __restrict__ qraw,
                            const float* __restrict__ kraw,
                            const float* __restrict__ vraw,
                            const int* __restrict__ pos_ids,
                            const float* __restrict__ shift,
                            _Float16* __restrict__ qh,
                            _Float16* __restrict__ kh,
                            _Float16* __restrict__ vT) {
    int idx = blockIdx.x * blockDim.x + threadIdx.x;   // over B*S*HID
    int d = idx & (DH - 1);
    int h = (idx >> 6) & (HN - 1);
    int s = (idx >> 10) & (SEQ - 1);
    int b = idx >> 21;

    float pos = (float)pos_ids[s];
    float inv = __expf(-0.28782313662425574f * (float)(d & 31)); // 10000^(-i/32)
    float sn, cs;
    __sincosf(pos * inv, &sn, &cs);

    int pair  = (d < 32) ? idx + 32 : idx - 32;
    float sgn = (d < 32) ? -1.f : 1.f;
    float sh  = shift[s];

    float qr = qraw[idx] * cs + sgn * qraw[pair] * sn + sh;
    float kr = kraw[idx] * cs + sgn * kraw[pair] * sn + sh;

    size_t ho = (size_t)((b * HN + h) * SEQ + s) * DH + d;
    qh[ho] = (_Float16)qr;
    kh[ho] = (_Float16)kr;
    vT[(size_t)((b * HN + h) * DH + d) * SEQ + s] = (_Float16)vraw[idx];
}

// ---------------------------------------------------------------------------
// 4. Fused attention: per (head, 16-query strip)
//    scores (WMMA -> LDS) -> softmax (LDS) -> write attn ONCE -> attn@V (LDS A,
//    global f16 B, WMMA) -> f16 ctx.  256 threads = 8 waves per workgroup.
// ---------------------------------------------------------------------------
__global__ void __launch_bounds__(256)
fused_attn_kernel(const _Float16* __restrict__ qh,
                  const _Float16* __restrict__ kh,
                  const _Float16* __restrict__ vT,
                  const int* __restrict__ amask,
                  float* __restrict__ attn,
                  _Float16* __restrict__ ctx) {
    extern __shared__ float smem[];
    float* sc    = smem + SM_SC;      // [16][SEQ]
    float* rstat = smem + SM_RSTAT;   // [16][17]
    float* rowst = smem + SM_ROWST;   // rowmax[16], rowinv[16]
    float* pacc  = smem + SM_PACC;    // [8][32][8]

    int wg   = blockIdx.x;            // head * 128 + strip
    int head = wg >> 7;
    int mt   = wg & 127;
    int b    = head / HN, h = head % HN;
    int i0   = mt * 16;

    int tid  = threadIdx.x;
    int wv   = tid >> 5;
    int lane = tid & 31;

    const _Float16* qhd = qh + (size_t)head * SEQ * DH;
    const _Float16* khd = kh + (size_t)head * SEQ * DH;
    const _Float16* vTh = vT + (size_t)head * DH * SEQ;

    // Q strip fragments, loaded once, reused for every score tile
    v16h aq0 = frag_a_f16(qhd, DH, i0 + (lane & 15), 0, lane);
    v16h aq1 = frag_a_f16(qhd, DH, i0 + (lane & 15), 32, lane);

    // ---- phase 1: score tiles into LDS (causal tiles only) ----
    for (int jt = wv; jt <= mt; jt += 8) {          // wave-uniform trip count
        int j0 = jt * 16;
        v16h b0 = frag_b_f16T(khd, DH, j0 + (lane & 15), 0, lane);
        v16h b1 = frag_b_f16T(khd, DH, j0 + (lane & 15), 32, lane);
        v8f acc = {};
        acc = wmma_f16(aq0, b0, acc);
        acc = wmma_f16(aq1, b1, acc);
        int ocol = j0 + (lane & 15);
        bool pad = (amask[b * SEQ + ocol] == 0);
        int rbase = (lane >> 4) << 3;
#pragma unroll
        for (int r = 0; r < 8; ++r) {
            int lr = rbase + r;                     // local row 0..15
            float v = acc[r] * 0.125f;              // 1/sqrt(64)
            if (pad || (ocol > i0 + lr)) v = NEG_INF;
            sc[lr * SEQ + ocol] = v;
        }
    }
    __syncthreads();

    // ---- phase 2: softmax over sc[:, 0..jlim) in LDS ----
    int jlim  = i0 + 16;
    int row   = tid & 15;
    int chunk = tid >> 4;                            // 16 threads per row
    float mx = NEG_INF;
    for (int j = chunk; j < jlim; j += 16) mx = fmaxf(mx, sc[row * SEQ + j]);
    rstat[row * 17 + chunk] = mx;
    __syncthreads();
    if (chunk == 0) {
        float m = rstat[row * 17];
        for (int c = 1; c < 16; ++c) m = fmaxf(m, rstat[row * 17 + c]);
        rowst[row] = m;
    }
    __syncthreads();
    float rm = rowst[row];
    float sm = 0.f;
    for (int j = chunk; j < jlim; j += 16) {
        float e = __expf(sc[row * SEQ + j] - rm);
        sc[row * SEQ + j] = e;
        sm += e;
    }
    rstat[row * 17 + chunk] = sm;
    __syncthreads();
    if (chunk == 0) {
        float s = 0.f;
        for (int c = 0; c < 16; ++c) s += rstat[row * 17 + c];
        rowst[16 + row] = 1.0f / s;
    }
    __syncthreads();
    float inv = rowst[16 + row];
    for (int j = chunk; j < jlim; j += 16) sc[row * SEQ + j] *= inv;
    __syncthreads();

    // ---- phase 3: single write of normalized attn strip ----
    float* aout = attn + (size_t)head * SEQ * SEQ + (size_t)i0 * SEQ;
    for (int idx = tid; idx < 16 * SEQ; idx += 256) {
        int r = idx >> 11;                           // /SEQ
        int j = idx & (SEQ - 1);
        aout[(size_t)r * SEQ + j] = (j < jlim) ? sc[r * SEQ + j] : 0.f;
    }

    // ---- phase 4: out strip = attn_strip @ V (A from LDS, causal-truncated K)
    // wave -> (n-tile = wv&3, K-half = wv>>2); halves reduced through LDS
    int n0   = (wv & 3) * 16;
    int kend = (jlim + 31) & ~31;
    int kmid = ((kend >> 1) + 31) & ~31;
    int k0 = (wv >> 2) ? kmid : 0;
    int k1 = (wv >> 2) ? kend : kmid;
    int arow_l = lane & 15;
    int bcol   = n0 + (lane & 15);
    v8f acc = {};
    for (int kk = k0; kk < k1; kk += 32) {
        if (kk + 32 < k1)
            __builtin_prefetch(vTh + (size_t)bcol * SEQ + kk + 32, 0, 1);
        v16h a  = frag_a_lds(sc, arow_l, kk, lane);
        v16h bb = frag_b_f16T(vTh, SEQ, bcol, kk, lane);
        acc = wmma_f16(a, bb, acc);
    }
#pragma unroll
    for (int r = 0; r < 8; ++r) pacc[(wv * 32 + lane) * 8 + r] = acc[r];
    __syncthreads();
    if (wv < 4) {
#pragma unroll
        for (int r = 0; r < 8; ++r) {
            float v = pacc[(wv * 32 + lane) * 8 + r] +
                      pacc[((wv + 4) * 32 + lane) * 8 + r];
            int orow = ((lane >> 4) << 3) + r;
            ctx[(size_t)(b * SEQ + i0 + orow) * HID + h * DH + n0 + (lane & 15)] =
                (_Float16)v;
        }
    }
}

// ---------------------------------------------------------------------------
// 5. out = ctx @ Wo^T  (register-blocked 16x64 per wave) -> f32 d_out
// ---------------------------------------------------------------------------
__global__ void outproj_kernel(const _Float16* __restrict__ ctx,
                               const float* __restrict__ Wo,
                               float* __restrict__ out) {
    int wave = blockIdx.x * (blockDim.x >> 5) + (threadIdx.x >> 5);
    int lane = threadIdx.x & 31;
    const int ngrp = HID / 64;                      // 16
    int mt = wave / ngrp, ng = wave % ngrp;
    int m0 = mt * 16, n0 = ng * 64;
    int arow = m0 + (lane & 15);
    v8f acc[4] = {};
    for (int kk = 0; kk < HID; kk += 32) {
        v16h a = frag_a_f16(ctx, HID, arow, kk, lane);
#pragma unroll
        for (int t = 0; t < 4; ++t) {
            v16h b = frag_b_f32T(Wo, HID, n0 + t * 16 + (lane & 15), kk, lane);
            acc[t] = wmma_f16(a, b, acc[t]);
        }
    }
    int orow = m0 + ((lane >> 4) << 3);
#pragma unroll
    for (int t = 0; t < 4; ++t) {
        int ocol = n0 + t * 16 + (lane & 15);
#pragma unroll
        for (int r = 0; r < 8; ++r)
            out[(size_t)(orow + r) * HID + ocol] = acc[t][r];
    }
}

// ---------------------------------------------------------------------------
extern "C" void kernel_launch(void* const* d_in, const int* in_sizes, int n_in,
                              void* d_out, int out_size, void* d_ws, size_t ws_size,
                              hipStream_t stream) {
    (void)in_sizes; (void)n_in; (void)out_size; (void)ws_size;

    const float* x     = (const float*)d_in[0];
    const int*   amask = (const int*)d_in[1];
    const int*   pos   = (const int*)d_in[2];
    const float* Wq    = (const float*)d_in[3];
    const float* Wk    = (const float*)d_in[4];
    const float* Wv    = (const float*)d_in[5];
    const float* Wo    = (const float*)d_in[6];
    const float* cbias = (const float*)d_in[7];

    float* out  = (float*)d_out;
    float* attn = out + (size_t)BATCH * SEQ * HID;   // tuple order: (out, attn)

    // workspace carve-up (~80 MB)
    char* ws = (char*)d_ws;
    float* shift = (float*)ws;            ws += 8192;
    float* qraw  = (float*)ws;            ws += (size_t)MROWS * HID * 4;
    float* kraw  = (float*)ws;            ws += (size_t)MROWS * HID * 4;
    float* vraw  = (float*)ws;            ws += (size_t)MROWS * HID * 4;
    _Float16* qh = (_Float16*)ws;         ws += (size_t)BATCH * HN * SEQ * DH * 2;
    _Float16* kh = (_Float16*)ws;         ws += (size_t)BATCH * HN * SEQ * DH * 2;
    _Float16* vT = (_Float16*)ws;         ws += (size_t)BATCH * HN * DH * SEQ * 2;
    _Float16* ctx = (_Float16*)ws;        ws += (size_t)MROWS * HID * 2;

    // 1. contextual-bias shift
    shift_kernel<<<SEQ, 256, 0, stream>>>(cbias, shift);

    // 2. fused QKV projection: 3 * 256 m-tiles * 16 n-groups waves, 4 waves/block
    proj_kernel<<<(3 * 256 * 16) / 4, 128, 0, stream>>>(x, Wq, Wk, Wv,
                                                        qraw, kraw, vraw);

    // 3. RoPE + shift + layout transform
    rope_kernel<<<(BATCH * SEQ * HID) / 256, 256, 0, stream>>>(
        qraw, kraw, vraw, pos, shift, qh, kh, vT);

    // 4. fused scores+softmax+attn@V: one workgroup per (head, 16-row strip)
    fused_attn_kernel<<<BATCH * HN * (SEQ / 16), 256, SM_BYTES, stream>>>(
        qh, kh, vT, amask, attn, ctx);

    // 5. output projection
    outproj_kernel<<<(256 * 16) / 4, 128, 0, stream>>>(ctx, Wo, out);
}
